// MoE_51900384805270
// MI455X (gfx1250) — matrix operand order repaired and checked by reading, not compiled, not run on hardware
//
#include <hip/hip_runtime.h>
#include <math.h>

// ---------------------------------------------------------------------------
// MoE sparse-dispatch for gfx1250 (wave32, WMMA bf16, async global->LDS)
// B=8192 D=1024 H=4096 E=8 K=2
// ---------------------------------------------------------------------------

typedef __bf16 bf16;
typedef __attribute__((ext_vector_type(16))) __bf16 v16bf;
typedef __attribute__((ext_vector_type(8)))  __bf16 v8bf;
typedef __attribute__((ext_vector_type(4)))  __bf16 v4bf;
typedef __attribute__((ext_vector_type(8)))  float  v8f;

#define NB      8192
#define ND      1024
#define NH      4096
#define NE      8
#define TM      128
#define TN      128
#define TK      32
#define LPITCH  40          // LDS row pitch in bf16 elems (80B -> conflict-free)
#define MAX_TILES 136       // 16384/128 + 8 (one partial tile per expert max)
#define MAXP    (MAX_TILES*TM)   // 17408 padded pair slots

// --------------------------- helpers ---------------------------------------

__device__ __forceinline__ v16bf ld_frag(const bf16* p, int hiOff) {
  union { v16bf v; v8bf h[2]; } u;
  u.h[0] = *(const v8bf*)(p);
  u.h[1] = *(const v8bf*)(p + hiOff);
  return u.v;
}

// per-lane 16B async copy global -> LDS (ASYNCcnt-tracked, no VGPR staging)
__device__ __forceinline__ void async_b128(uint32_t ldsOff, const char* gptr) {
  asm volatile("global_load_async_to_lds_b128 %0, %1, off"
               :: "v"(ldsOff), "v"((uint64_t)(uintptr_t)gptr)
               : "memory");
}
__device__ __forceinline__ void wait_async_le(int n) {
  if (n == 0) asm volatile("s_wait_asynccnt 0x0" ::: "memory");
  else        asm volatile("s_wait_asynccnt 0x4" ::: "memory");
}

// --------------------------- conversion kernels ----------------------------

__global__ __launch_bounds__(256) void cvt_x_kernel(const float* __restrict__ in,
                                                    bf16* __restrict__ out) {
  int i = blockIdx.x * blockDim.x + threadIdx.x;       // i < NB*ND/4
  float4 v = reinterpret_cast<const float4*>(in)[i];
  v4bf o = { (bf16)v.x, (bf16)v.y, (bf16)v.z, (bf16)v.w };
  reinterpret_cast<v4bf*>(out)[i] = o;
}

// in: [E][R][C] f32 -> out: [E][C][R] bf16
__global__ __launch_bounds__(256) void transpose_bf16_kernel(
    const float* __restrict__ in, bf16* __restrict__ out, int R, int C) {
  __shared__ float tile[32][33];
  int e = blockIdx.z;
  const float* src = in + (size_t)e * R * C;
  bf16* dst = out + (size_t)e * R * C;
  int c0 = blockIdx.x * 32, r0 = blockIdx.y * 32;
  int tx = threadIdx.x, ty = threadIdx.y;
#pragma unroll
  for (int i = 0; i < 32; i += 8)
    tile[ty + i][tx] = src[(size_t)(r0 + ty + i) * C + (c0 + tx)];
  __syncthreads();
#pragma unroll
  for (int i = 0; i < 32; i += 8)
    dst[(size_t)(c0 + ty + i) * R + (r0 + tx)] = (bf16)tile[tx][ty + i];
}

// --------------------------- router ----------------------------------------

__global__ __launch_bounds__(256) void router_kernel(
    const float* __restrict__ x, const float* __restrict__ Wr,
    const float* __restrict__ br,
    int* __restrict__ topE, float* __restrict__ topW, int* __restrict__ counts) {
  int wave = threadIdx.x >> 5;
  int lane = threadIdx.x & 31;
  int b = blockIdx.x * 8 + wave;
  if (b >= NB) return;
  float acc[8] = {0.f,0.f,0.f,0.f,0.f,0.f,0.f,0.f};
  const float* xr = x + (size_t)b * ND;
  for (int d = lane; d < ND; d += 32) {
    float xv = xr[d];
    const float4 w0 = *reinterpret_cast<const float4*>(Wr + d * 8);
    const float4 w1 = *reinterpret_cast<const float4*>(Wr + d * 8 + 4);
    acc[0] += xv * w0.x; acc[1] += xv * w0.y; acc[2] += xv * w0.z; acc[3] += xv * w0.w;
    acc[4] += xv * w1.x; acc[5] += xv * w1.y; acc[6] += xv * w1.z; acc[7] += xv * w1.w;
  }
#pragma unroll
  for (int e = 0; e < 8; e++)
#pragma unroll
    for (int off = 16; off > 0; off >>= 1)
      acc[e] += __shfl_xor(acc[e], off, 32);
  if (lane == 0) {
    float l[8];
#pragma unroll
    for (int e = 0; e < 8; e++) l[e] = acc[e] + br[e];
    int e0 = 0;
#pragma unroll
    for (int e = 1; e < 8; e++) if (l[e] > l[e0]) e0 = e;       // ties: lower idx
    int e1 = (e0 == 0) ? 1 : 0;
#pragma unroll
    for (int e = 0; e < 8; e++) if (e != e0 && l[e] > l[e1]) e1 = e;
    float p1 = __expf(l[e1] - l[e0]);       // p0 = 1
    float s = 1.0f + p1;
    topE[2 * b + 0] = e0; topW[2 * b + 0] = 1.0f / s;
    topE[2 * b + 1] = e1; topW[2 * b + 1] = p1 / s;
    atomicAdd(&counts[e0], 1);
    atomicAdd(&counts[e1], 1);
  }
}

// --------------------------- segment scan ----------------------------------

__global__ void scan_kernel(const int* __restrict__ counts, int* __restrict__ segBase,
                            int* __restrict__ tileExp, int* __restrict__ tileRow) {
  if (threadIdx.x != 0 || blockIdx.x != 0) return;
  int base = 0, t = 0;
  for (int e = 0; e < NE; e++) {
    segBase[e] = base;
    int nt = (counts[e] + TM - 1) >> 7;
    for (int i = 0; i < nt; i++) { tileExp[t] = e; tileRow[t] = base + i * TM; t++; }
    base += nt << 7;
  }
  for (; t < MAX_TILES; t++) tileExp[t] = -1;
}

// --------------------------- scatter ---------------------------------------

__global__ __launch_bounds__(256) void scatter_kernel(
    const int* __restrict__ topE, const float* __restrict__ topW,
    const int* __restrict__ segBase, int* __restrict__ cursor,
    int* __restrict__ pairTok, float* __restrict__ pairW) {
  int b = blockIdx.x * blockDim.x + threadIdx.x;
  if (b >= NB) return;
#pragma unroll
  for (int k = 0; k < 2; k++) {
    int e = topE[2 * b + k];
    int pos = atomicAdd(&cursor[e], 1);
    int idx = segBase[e] + pos;
    pairTok[idx] = b;
    pairW[idx] = topW[2 * b + k];
  }
}

// --------------------------- GEMM1: H = gelu(Xg @ W1 + b1) -----------------

__global__ __launch_bounds__(256) void gemm1_kernel(
    const bf16* __restrict__ xb,    // [B, D]
    const bf16* __restrict__ W1b,   // [E, H, D]  (K-contiguous)
    const float* __restrict__ b1,   // [E, H]
    const int* __restrict__ tileExp, const int* __restrict__ tileRow,
    const int* __restrict__ pairTok,
    bf16* __restrict__ Hact) {      // [MAXP, H]
  int t = blockIdx.x;
  int e = tileExp[t];
  if (e < 0) return;
  int rowStart = tileRow[t];
  int h0 = blockIdx.y * TN;

  __shared__ __align__(16) bf16 lA[2][TM * LPITCH];
  __shared__ __align__(16) bf16 lB[2][TN * LPITCH];
  __shared__ int lTok[TM];

  int tid = threadIdx.x;
  int lane = tid & 31;
  int wave = tid >> 5;
  int wm = wave & 1;    // 0..1 over M (64 rows each)
  int wn = wave >> 1;   // 0..3 over N (32 cols each)

  if (tid < TM) lTok[tid] = pairTok[rowStart + tid];
  __syncthreads();

  const bf16* Bsrc = W1b + ((size_t)e * NH + h0) * ND;

  // --- per-lane async-copy assignment: wave w stages rows [w*16, w*16+16) ---
  int rA = wave * 16 + (lane >> 2);     // first row (second is +8)
  int seg = (lane & 3) * 16;            // byte offset within the 64B row slice
  int tokA0 = lTok[rA];     if (tokA0 < 0) tokA0 = 0;
  int tokA1 = lTok[rA + 8]; if (tokA1 < 0) tokA1 = 0;
  const char* gA0 = (const char*)(xb + (size_t)tokA0 * ND) + seg;
  const char* gA1 = (const char*)(xb + (size_t)tokA1 * ND) + seg;
  const char* gB0 = (const char*)(Bsrc + (size_t)rA * ND) + seg;
  const char* gB1 = (const char*)(Bsrc + (size_t)(rA + 8) * ND) + seg;
  uint32_t dA0[2], dA1[2], dB0[2], dB1[2];
#pragma unroll
  for (int s = 0; s < 2; s++) {
    dA0[s] = (uint32_t)(size_t)((const char*)&lA[s][rA * LPITCH] + seg);
    dA1[s] = (uint32_t)(size_t)((const char*)&lA[s][(rA + 8) * LPITCH] + seg);
    dB0[s] = (uint32_t)(size_t)((const char*)&lB[s][rA * LPITCH] + seg);
    dB1[s] = (uint32_t)(size_t)((const char*)&lB[s][(rA + 8) * LPITCH] + seg);
  }
  auto issue = [&](int buf, int k0) {
    size_t kb = (size_t)k0 * 2;
    async_b128(dA0[buf], gA0 + kb);
    async_b128(dA1[buf], gA1 + kb);
    async_b128(dB0[buf], gB0 + kb);
    async_b128(dB1[buf], gB1 + kb);
  };

  v8f acc[4][2] = {};
  int r15 = lane & 15;
  int aK = (lane >> 4) * 16;
  int bK = (lane >> 4) * 16;
  const int NK = ND / TK;

  issue(0, 0);
  for (int i = 0; i < NK; i++) {
    int k1 = (i + 1) * TK;
    if (k1 < ND) issue((i + 1) & 1, k1);
    wait_async_le(k1 < ND ? 4 : 0);     // own stage-i transfers retired
    __syncthreads();                     // all waves' stage-i LDS writes visible
    const bf16* bufA = lA[i & 1];
    const bf16* bufB = lB[i & 1];
    v16bf af[4], bfr[2];
#pragma unroll
    for (int m = 0; m < 4; m++)
      af[m] = ld_frag(bufA + (wm * 64 + m * 16 + r15) * LPITCH + aK, 8);
#pragma unroll
    for (int j = 0; j < 2; j++)
      bfr[j] = ld_frag(bufB + (wn * 32 + j * 16 + r15) * LPITCH + bK, 8);
#pragma unroll
    for (int m = 0; m < 4; m++)
#pragma unroll
      for (int j = 0; j < 2; j++)
        acc[m][j] = __builtin_amdgcn_wmma_f32_16x16x32_bf16(
            false, af[m], false, bfr[j], (short)0, acc[m][j], false, false);
    __syncthreads();                     // buffer reuse safety for stage i+2
  }

  int mhalf = (lane >> 4) * 8;
#pragma unroll
  for (int m = 0; m < 4; m++) {
#pragma unroll
    for (int j = 0; j < 2; j++) {
      int col = h0 + wn * 32 + j * 16 + r15;
      float bias = b1[(size_t)e * NH + col];
#pragma unroll
      for (int r = 0; r < 8; r++) {
        int mrow = wm * 64 + m * 16 + mhalf + r;
        int tok = lTok[mrow];
        if (tok >= 0) {
          float v = acc[m][j][r] + bias;
          float g = 0.5f * v * (1.0f + erff(v * 0.70710678118654752f));
          Hact[(size_t)(rowStart + mrow) * NH + col] = (bf16)g;
        }
      }
    }
  }
}

// --------------------------- GEMM2: out += w * (H @ W2 + b2) ---------------

__global__ __launch_bounds__(256) void gemm2_kernel(
    const bf16* __restrict__ Hact,  // [MAXP, H]
    const bf16* __restrict__ W2b,   // [E, D, H]  (K-contiguous)
    const float* __restrict__ b2,   // [E, D]
    const int* __restrict__ tileExp, const int* __restrict__ tileRow,
    const int* __restrict__ pairTok, const float* __restrict__ pairW,
    float* __restrict__ out) {      // [B, D]
  int t = blockIdx.x;
  int e = tileExp[t];
  if (e < 0) return;
  int rowStart = tileRow[t];
  int d0 = blockIdx.y * TN;

  __shared__ __align__(16) bf16 lA[2][TM * LPITCH];
  __shared__ __align__(16) bf16 lB[2][TN * LPITCH];
  __shared__ int lTok[TM];
  __shared__ float lW[TM];

  int tid = threadIdx.x;
  int lane = tid & 31;
  int wave = tid >> 5;
  int wm = wave & 1;
  int wn = wave >> 1;

  if (tid < TM) {
    lTok[tid] = pairTok[rowStart + tid];
    lW[tid] = pairW[rowStart + tid];
  }
  __syncthreads();

  const bf16* Asrc = Hact + (size_t)rowStart * NH;
  const bf16* Bsrc = W2b + ((size_t)e * ND + d0) * NH;

  int rA = wave * 16 + (lane >> 2);
  int seg = (lane & 3) * 16;
  const char* gA0 = (const char*)(Asrc + (size_t)rA * NH) + seg;
  const char* gA1 = (const char*)(Asrc + (size_t)(rA + 8) * NH) + seg;
  const char* gB0 = (const char*)(Bsrc + (size_t)rA * NH) + seg;
  const char* gB1 = (const char*)(Bsrc + (size_t)(rA + 8) * NH) + seg;
  uint32_t dA0[2], dA1[2], dB0[2], dB1[2];
#pragma unroll
  for (int s = 0; s < 2; s++) {
    dA0[s] = (uint32_t)(size_t)((const char*)&lA[s][rA * LPITCH] + seg);
    dA1[s] = (uint32_t)(size_t)((const char*)&lA[s][(rA + 8) * LPITCH] + seg);
    dB0[s] = (uint32_t)(size_t)((const char*)&lB[s][rA * LPITCH] + seg);
    dB1[s] = (uint32_t)(size_t)((const char*)&lB[s][(rA + 8) * LPITCH] + seg);
  }
  auto issue = [&](int buf, int k0) {
    size_t kb = (size_t)k0 * 2;
    async_b128(dA0[buf], gA0 + kb);
    async_b128(dA1[buf], gA1 + kb);
    async_b128(dB0[buf], gB0 + kb);
    async_b128(dB1[buf], gB1 + kb);
  };

  v8f acc[4][2] = {};
  int r15 = lane & 15;
  int aK = (lane >> 4) * 16;
  int bK = (lane >> 4) * 16;
  const int NK = NH / TK;

  issue(0, 0);
  for (int i = 0; i < NK; i++) {
    int k1 = (i + 1) * TK;
    if (k1 < NH) issue((i + 1) & 1, k1);
    wait_async_le(k1 < NH ? 4 : 0);
    __syncthreads();
    const bf16* bufA = lA[i & 1];
    const bf16* bufB = lB[i & 1];
    v16bf af[4], bfr[2];
#pragma unroll
    for (int m = 0; m < 4; m++)
      af[m] = ld_frag(bufA + (wm * 64 + m * 16 + r15) * LPITCH + aK, 8);
#pragma unroll
    for (int j = 0; j < 2; j++)
      bfr[j] = ld_frag(bufB + (wn * 32 + j * 16 + r15) * LPITCH + bK, 8);
#pragma unroll
    for (int m = 0; m < 4; m++)
#pragma unroll
      for (int j = 0; j < 2; j++)
        acc[m][j] = __builtin_amdgcn_wmma_f32_16x16x32_bf16(
            false, af[m], false, bfr[j], (short)0, acc[m][j], false, false);
    __syncthreads();
  }

  int mhalf = (lane >> 4) * 8;
#pragma unroll
  for (int m = 0; m < 4; m++) {
#pragma unroll
    for (int j = 0; j < 2; j++) {
      int col = d0 + wn * 32 + j * 16 + r15;
      float bias = b2[(size_t)e * ND + col];
#pragma unroll
      for (int r = 0; r < 8; r++) {
        int mrow = wm * 64 + m * 16 + mhalf + r;
        int tok = lTok[mrow];
        if (tok >= 0) {
          float v = (acc[m][j][r] + bias) * lW[mrow];
          atomicAdd(&out[(size_t)tok * ND + col], v);
        }
      }
    }
  }
}

// --------------------------- host launch -----------------------------------

static inline size_t alignUp(size_t v) { return (v + 255) & ~(size_t)255; }

extern "C" void kernel_launch(void* const* d_in, const int* in_sizes, int n_in,
                              void* d_out, int out_size, void* d_ws, size_t ws_size,
                              hipStream_t stream) {
  (void)in_sizes; (void)n_in; (void)out_size; (void)ws_size;
  const float* x  = (const float*)d_in[0];
  const float* Wr = (const float*)d_in[1];
  const float* br = (const float*)d_in[2];
  const float* W1 = (const float*)d_in[3];
  const float* b1 = (const float*)d_in[4];
  const float* W2 = (const float*)d_in[5];
  const float* b2 = (const float*)d_in[6];
  float* out = (float*)d_out;

  // workspace carve-up
  char* p = (char*)d_ws;
  bf16*  xb      = (bf16*)p;  p += alignUp((size_t)NB * ND * 2);
  bf16*  W1b     = (bf16*)p;  p += alignUp((size_t)NE * NH * ND * 2);
  bf16*  W2b     = (bf16*)p;  p += alignUp((size_t)NE * ND * NH * 2);
  bf16*  Hact    = (bf16*)p;  p += alignUp((size_t)MAXP * NH * 2);
  int*   pairTok = (int*)p;   p += alignUp((size_t)MAXP * 4);
  float* pairW   = (float*)p; p += alignUp((size_t)MAXP * 4);
  int*   topE    = (int*)p;   p += alignUp((size_t)NB * 2 * 4);
  float* topW    = (float*)p; p += alignUp((size_t)NB * 2 * 4);
  int*   counts  = (int*)p;   p += alignUp(NE * 4);
  int*   cursor  = (int*)p;   p += alignUp(NE * 4);
  int*   segBase = (int*)p;   p += alignUp(NE * 4);
  int*   tileExp = (int*)p;   p += alignUp(MAX_TILES * 4);
  int*   tileRow = (int*)p;   p += alignUp(MAX_TILES * 4);

  hipMemsetAsync(out, 0, (size_t)NB * ND * 4, stream);
  hipMemsetAsync(counts, 0, NE * 4, stream);
  hipMemsetAsync(cursor, 0, NE * 4, stream);
  hipMemsetAsync(pairTok, 0xFF, (size_t)MAXP * 4, stream);   // all tokens = -1

  cvt_x_kernel<<<(NB * ND / 4) / 256, 256, 0, stream>>>(x, xb);
  transpose_bf16_kernel<<<dim3(NH / 32, ND / 32, NE), dim3(32, 8), 0, stream>>>(
      W1, W1b, ND, NH);   // [E,D,H] -> [E,H,D]
  transpose_bf16_kernel<<<dim3(ND / 32, NH / 32, NE), dim3(32, 8), 0, stream>>>(
      W2, W2b, NH, ND);   // [E,H,D] -> [E,D,H]

  router_kernel<<<NB / 8, 256, 0, stream>>>(x, Wr, br, topE, topW, counts);
  scan_kernel<<<1, 1, 0, stream>>>(counts, segBase, tileExp, tileRow);
  scatter_kernel<<<NB / 256, 256, 0, stream>>>(topE, topW, segBase, cursor,
                                               pairTok, pairW);

  gemm1_kernel<<<dim3(MAX_TILES, NH / TN), 256, 0, stream>>>(
      xb, W1b, b1, tileExp, tileRow, pairTok, Hact);
  gemm2_kernel<<<dim3(MAX_TILES, ND / TN), 256, 0, stream>>>(
      Hact, W2b, b2, tileExp, tileRow, pairTok, pairW, out);
}
